// PointNetFeat_16544214024681
// MI455X (gfx1250) — matrix operand order repaired and checked by reading, not compiled, not run on hardware
//
#include <hip/hip_runtime.h>

// PointNetFeat forward for MI455X (gfx1250, wave32, WMMA + TDM).
// Compute-bound (~110 GFLOP, hot data L2-resident): ci>=32 GEMMs via
// v_wmma_f32_16x16x32_f16 (f32 accumulate), BN folded to scale/shift,
// channel-max fused (never materialize (B,1024,N)), tile ingest via
// tensor_load_to_lds (TDM) with LDS row padding.

#define USE_TDM 1

typedef __attribute__((ext_vector_type(16))) _Float16 v16h;
typedef __attribute__((ext_vector_type(8)))  float    v8f;
typedef __attribute__((ext_vector_type(4)))  unsigned tdm_u32x4;
typedef __attribute__((ext_vector_type(8)))  int      tdm_i32x8;
typedef __attribute__((ext_vector_type(4)))  int      tdm_i32x4;

#define EPS_BN 1e-5f
#define BATCH 32
#define NPTS 4096
#define NT 128              // points per workgroup tile
#define TPB 256             // 8 waves, each owns 16 points
#define STR64 72            // LDS row stride (f16) for 64-ch act: 144B, 16B-aligned, conflict-free
#define STR128 136          // 272B rows for 128-ch act
#define NEG_FILL (-3.402823466e38f)

// monotone float <-> uint mapping so atomicMax(uint) == float max; 0 is identity
__device__ __forceinline__ unsigned f2ord(float f) {
  unsigned i = __float_as_uint(f);
  return (i & 0x80000000u) ? ~i : (i | 0x80000000u);
}
__device__ __forceinline__ float ord2f(unsigned u) {
  unsigned i = (u & 0x80000000u) ? (u & 0x7FFFFFFFu) : ~u;
  return __uint_as_float(i);
}

// ---- TDM: rows x 128B contiguous global -> LDS with 144B row stride --------
// D# per CDNA5 ISA: pad_enable, pad_interval=4 (32 dwords), pad_amount=3 (4 dwords).
// Toolchain here is the clang-23 lane: 6-arg builtin
//   (uint32x4 g0, int32x8 g1, int32x4 g2, int32x4 g3, int32x8 extra, i32 cpol)
#if USE_TDM && __has_builtin(__builtin_amdgcn_tensor_load_to_lds)
#define HAVE_TDM 1
__device__ __forceinline__ void tdm_load_tile_128B(void* lds_dst, const void* gsrc, int rows) {
  unsigned lds = (unsigned)(unsigned long long)lds_dst;   // flat addr low 32 = LDS offset
  unsigned long long ga = (unsigned long long)gsrc;
  tdm_u32x4 g0 = { 1u,                                    // count=1
                   lds,                                   // lds_addr [63:32]
                   (unsigned)(ga & 0xFFFFFFFFull),        // global_addr [95:64]
                   ((unsigned)((ga >> 32) & 0x1FFFFFFull)) | (2u << 30) };  // [120:96] | type=2
  tdm_i32x8 g1 = { (int)((2u << 16) | (1u << 20) | (4u << 22) | (3u << 25)),
                   (int)(32u << 16),                      // tensor_dim0 = 32 dwords @bit48
                   (int)(((unsigned)rows) << 16),         // tensor_dim1 = rows   @bit80
                   (int)(32u << 16),                      // tile_dim0 = 32       @bit112
                   rows,                                  // tile_dim1            @bit128
                   32, 0, 0 };                            // tensor_dim0_stride=32 @bit160
  tdm_i32x4 z4 = { 0, 0, 0, 0 };
  tdm_i32x8 z8 = { 0, 0, 0, 0, 0, 0, 0, 0 };
  __builtin_amdgcn_tensor_load_to_lds(g0, g1, z4, z4, z8, 0);
}
#else
#define HAVE_TDM 0
#endif

// ---- WMMA fragment loaders -------------------------------------------------
// A: 16x32 f16 tile of W[COUT][CIN] row-major (M=co, K=ci).
template<int CIN>
__device__ __forceinline__ v16h load_a_w(const _Float16* W, int co_base, int ci_base, int lane) {
  int m  = co_base + (lane & 15);
  int hh = lane >> 4;
  const _Float16* row = W + m * CIN + ci_base;
  v16h a;
#pragma unroll
  for (int j = 0; j < 8; ++j) {
    int k = (j < 4) ? (2*j + 8*hh) : (2*(j-4) + 16 + 8*hh);
    a[2*j]   = row[k];
    a[2*j+1] = row[k+1];
  }
  return a;
}

// A tile from per-batch f32 transform T stored [c][k] (64x64): A[m=k][ci=c]=T[c][k]
__device__ __forceinline__ v16h load_a_t64(const float* T, int k_base, int c_base, int lane) {
  int m  = k_base + (lane & 15);
  int hh = lane >> 4;
  v16h a;
#pragma unroll
  for (int j = 0; j < 8; ++j) {
    int c = c_base + ((j < 4) ? (2*j + 8*hh) : (2*(j-4) + 16 + 8*hh));
    a[2*j]   = (_Float16)T[c*64 + m];
    a[2*j+1] = (_Float16)T[(c+1)*64 + m];
  }
  return a;
}

// B: 32x16 f16 tile from LDS activations stored transposed actT[n][ci] (stride STR).
template<int STR>
__device__ __forceinline__ v16h load_b_act(const _Float16* actT, int n_base, int ci_base, int lane) {
  const _Float16* row = actT + (n_base + (lane & 15)) * STR + ci_base + 8*(lane >> 4);
  v16h b;
#pragma unroll
  for (int j = 0; j < 16; ++j) b[j] = row[(j < 8) ? j : (j + 8)];
  return b;
}

// ---- Pointwise-conv layer via WMMA: actOut = relu?(sc*acc + sh) ------------
template<int CIN, int COUT, int ISTR, int OSTR, bool RELU>
__device__ __forceinline__ void conv_layer(const _Float16* actIn, _Float16* actOut,
    const _Float16* W, const float* sc, const float* sh, int wave, int lane) {
  int n_base = wave * 16;
  int col = lane & 15, hh = lane >> 4;
  v16h bf[CIN/32];
#pragma unroll
  for (int ks = 0; ks < CIN/32; ++ks) bf[ks] = load_b_act<ISTR>(actIn, n_base, ks*32, lane);
#pragma unroll 1
  for (int cot = 0; cot < COUT/16; ++cot) {
    v8f acc = {};
#pragma unroll
    for (int ks = 0; ks < CIN/32; ++ks) {
      v16h a = load_a_w<CIN>(W, cot*16, ks*32, lane);
      acc = __builtin_amdgcn_wmma_f32_16x16x32_f16(false, a, false, bf[ks], (short)0, acc, false, false);
    }
    int co0 = cot*16 + 8*hh;                     // D VGPR r <-> m = r + 8*half
    _Float16* out = actOut + (n_base + col)*OSTR + co0;
#pragma unroll
    for (int r = 0; r < 8; ++r) {
      float y = fmaf(acc[r], sc[co0+r], sh[co0+r]);
      if (RELU) y = fmaxf(y, 0.f);
      out[r] = (_Float16)y;
    }
  }
}

// ---- Final 1024-ch layer fused with (optionally masked) max over points ----
template<int CIN, int ISTR, bool RELU, bool MASK>
__device__ __forceinline__ void conv_final_max(const _Float16* actIn,
    const _Float16* W, const float* sc, const float* sh,
    unsigned* wsmax, int b, int n_tile0, int vp, int wave, int lane) {
  int n_base = wave * 16;
  int col = lane & 15, hh = lane >> 4;
  bool valid = true;
  if (MASK) { int n = n_tile0 + n_base + col; valid = (n < vp) || (vp == 0); }
  v16h bf[CIN/32];
#pragma unroll
  for (int ks = 0; ks < CIN/32; ++ks) bf[ks] = load_b_act<ISTR>(actIn, n_base, ks*32, lane);
#pragma unroll 1
  for (int cot = 0; cot < 64; ++cot) {   // COUT = 1024
    v8f acc = {};
#pragma unroll
    for (int ks = 0; ks < CIN/32; ++ks) {
      v16h a = load_a_w<CIN>(W, cot*16, ks*32, lane);
      acc = __builtin_amdgcn_wmma_f32_16x16x32_f16(false, a, false, bf[ks], (short)0, acc, false, false);
    }
    int co0 = cot*16 + 8*hh;
#pragma unroll
    for (int r = 0; r < 8; ++r) {
      float y = fmaf(acc[r], sc[co0+r], sh[co0+r]);
      if (RELU) y = fmaxf(y, 0.f);
      y = valid ? y : NEG_FILL;
#pragma unroll
      for (int off = 1; off < 16; off <<= 1) y = fmaxf(y, __shfl_xor(y, off, 32));
      if (col == 0) atomicMax(&wsmax[b*1024 + co0 + r], f2ord(y));
    }
  }
}

// ---- Kernel: STN(3) conv trunk 3->64->128->1024 + max ----------------------
struct Stn3Args {
  const float* x;
  const float* c1w; const float* sc1; const float* sh1;
  const _Float16* c2w; const float* sc2; const float* sh2;
  const _Float16* c3w; const float* sc3; const float* sh3;
  unsigned* wsmax;
};

__global__ __launch_bounds__(TPB) void stn3_conv_kernel(Stn3Args A) {
  __shared__ __align__(16) _Float16 act64[NT*STR64];
  __shared__ __align__(16) _Float16 act128[NT*STR128];
  int b  = blockIdx.x / (NPTS/NT);
  int n0 = (blockIdx.x % (NPTS/NT)) * NT;
  int wave = threadIdx.x >> 5, lane = threadIdx.x & 31;
  int col = lane & 15, hh = lane >> 4;
  int nl = wave*16 + col;
  int ng = n0 + nl;
  // L1: 3->64, K=3 -> VALU; lane (col,hh) fills channels [32h,32h+32) of its point.
  float x0 = A.x[(b*3+0)*NPTS + ng];
  float x1 = A.x[(b*3+1)*NPTS + ng];
  float x2 = A.x[(b*3+2)*NPTS + ng];
  _Float16* orow = act64 + nl*STR64 + hh*32;
#pragma unroll 1
  for (int cc = 0; cc < 32; ++cc) {
    int c = hh*32 + cc;
    float y = A.c1w[c*3+0]*x0 + A.c1w[c*3+1]*x1 + A.c1w[c*3+2]*x2;
    orow[cc] = (_Float16)fmaxf(fmaf(y, A.sc1[c], A.sh1[c]), 0.f);
  }
  conv_layer<64,128,STR64,STR128,true>(act64, act128, A.c2w, A.sc2, A.sh2, wave, lane);
  conv_final_max<128,STR128,true,false>(act128, A.c3w, A.sc3, A.sh3, A.wsmax, b, n0, 0, wave, lane);
}

// ---- shared tile ingest (h1 (B,N,64) f16 -> LDS actA, row stride STR64) ----
__device__ __forceinline__ void load_h1_tile(_Float16* actA, const _Float16* h1,
                                             int b, int n0, int wave, int lane) {
#if HAVE_TDM
  if (wave == 0) {
    tdm_load_tile_128B(actA, h1 + ((size_t)b*NPTS + n0)*64, NT);
    __builtin_amdgcn_s_wait_tensorcnt(0);
  }
  __syncthreads();
#else
  int col = lane & 15, hh = lane >> 4;
  int nl = wave*16 + col;
  const _Float16* src = h1 + ((size_t)b*NPTS + n0 + nl)*64 + hh*32;
  _Float16* dst = actA + nl*STR64 + hh*32;
#pragma unroll
  for (int j = 0; j < 32; ++j) dst[j] = src[j];
#endif
}

// ---- Kernel: FSTN conv trunk 64->64->128->1024 + max -----------------------
struct FstnArgs {
  const _Float16* h1;
  const _Float16* c1w; const float* sc1; const float* sh1;
  const _Float16* c2w; const float* sc2; const float* sh2;
  const _Float16* c3w; const float* sc3; const float* sh3;
  unsigned* wsmax;
};

__global__ __launch_bounds__(TPB) void fstn_conv_kernel(FstnArgs A) {
  __shared__ __align__(16) _Float16 actA[NT*STR64];
  __shared__ __align__(16) _Float16 actB[NT*STR64];
  __shared__ __align__(16) _Float16 act128[NT*STR128];
  int b  = blockIdx.x / (NPTS/NT);
  int n0 = (blockIdx.x % (NPTS/NT)) * NT;
  int wave = threadIdx.x >> 5, lane = threadIdx.x & 31;
  load_h1_tile(actA, A.h1, b, n0, wave, lane);
  conv_layer<64,64,STR64,STR64,true>(actA, actB, A.c1w, A.sc1, A.sh1, wave, lane);
  conv_layer<64,128,STR64,STR128,true>(actB, act128, A.c2w, A.sc2, A.sh2, wave, lane);
  conv_final_max<128,STR128,true,false>(act128, A.c3w, A.sc3, A.sh3, A.wsmax, b, n0, 0, wave, lane);
}

// ---- Kernel: feature-transform + c2 + c3 + masked max ----------------------
struct MainArgs {
  const _Float16* h1;
  const float* tfeat;    // (B,64,64) f32 (in d_out)
  const _Float16* c2w; const float* sc2; const float* sh2;
  const _Float16* c3w; const float* sc3; const float* sh3;
  const int* vp;
  unsigned* wsmax;
};

__global__ __launch_bounds__(TPB) void main_conv_kernel(MainArgs A) {
  __shared__ __align__(16) _Float16 actA[NT*STR64];
  __shared__ __align__(16) _Float16 actB[NT*STR64];
  __shared__ __align__(16) _Float16 act128[NT*STR128];
  int b  = blockIdx.x / (NPTS/NT);
  int n0 = (blockIdx.x % (NPTS/NT)) * NT;
  int wave = threadIdx.x >> 5, lane = threadIdx.x & 31;
  int col = lane & 15, hh = lane >> 4;
  int vp = A.vp[b];
  load_h1_tile(actA, A.h1, b, n0, wave, lane);
  // feature transform via WMMA: actB[n][k] = sum_c actA[n][c] * T[c][k]
  const float* T = A.tfeat + b*4096;
  int n_base = wave*16;
  v16h bf[2];
#pragma unroll
  for (int ks = 0; ks < 2; ++ks) bf[ks] = load_b_act<STR64>(actA, n_base, ks*32, lane);
#pragma unroll 1
  for (int kt = 0; kt < 4; ++kt) {
    v8f acc = {};
#pragma unroll
    for (int ks = 0; ks < 2; ++ks) {
      v16h a = load_a_t64(T, kt*16, ks*32, lane);
      acc = __builtin_amdgcn_wmma_f32_16x16x32_f16(false, a, false, bf[ks], (short)0, acc, false, false);
    }
    _Float16* out = actB + (n_base + col)*STR64 + kt*16 + 8*hh;
#pragma unroll
    for (int r = 0; r < 8; ++r) out[r] = (_Float16)acc[r];
  }
  conv_layer<64,128,STR64,STR128,true>(actB, act128, A.c2w, A.sc2, A.sh2, wave, lane);
  conv_final_max<128,STR128,false,true>(act128, A.c3w, A.sc3, A.sh3, A.wsmax, b, n0, vp, wave, lane);
}

// ---- Kernel: STN FC head 1024->512->256->K*K (+identity) -------------------
struct FcArgs {
  const unsigned* zmax;
  const float* f1w; const float* f1b;
  const float* f2w; const float* f2b;
  const float* f3w; const float* f3b;
  const float* bn4g; const float* bn4b; const float* bn4m; const float* bn4v;
  const float* bn5g; const float* bn5b; const float* bn5m; const float* bn5v;
  float* out;
};

template<int K>
__global__ __launch_bounds__(256) void stn_fc_kernel(FcArgs A) {
  __shared__ float z[1024];
  __shared__ float h1[512];
  __shared__ float h2[256];
  int b = blockIdx.x, t = threadIdx.x;
  for (int i = t; i < 1024; i += 256) z[i] = ord2f(A.zmax[b*1024 + i]);
  __syncthreads();
  for (int o = t; o < 512; o += 256) {
    float acc = A.f1b[o];
    const float* w = A.f1w + o*1024;
    for (int i = 0; i < 1024; ++i) acc += w[i]*z[i];
    float s = A.bn4g[o]*rsqrtf(A.bn4v[o]+EPS_BN);
    h1[o] = fmaxf(s*(acc - A.bn4m[o]) + A.bn4b[o], 0.f);
  }
  __syncthreads();
  for (int o = t; o < 256; o += 256) {
    float acc = A.f2b[o];
    const float* w = A.f2w + o*512;
    for (int i = 0; i < 512; ++i) acc += w[i]*h1[i];
    float s = A.bn5g[o]*rsqrtf(A.bn5v[o]+EPS_BN);
    h2[o] = fmaxf(s*(acc - A.bn5m[o]) + A.bn5b[o], 0.f);
  }
  __syncthreads();
  for (int o = t; o < K*K; o += 256) {
    float acc = A.f3b[o];
    const float* w = A.f3w + o*256;
    for (int i = 0; i < 256; ++i) acc += w[i]*h2[i];
    if (o / K == o % K) acc += 1.0f;
    A.out[(size_t)b*K*K + o] = acc;
  }
}

// ---- Kernel: apply 3x3 trans + c1(3->64)+bn+relu -> h1 (B,N,64) f16 --------
__global__ __launch_bounds__(256) void apply3_c1_kernel(
    const float* x, const float* trans, const float* w,
    const float* sc, const float* sh, _Float16* h1out) {
  int idx = blockIdx.x*256 + threadIdx.x;          // exact grid: B*N threads
  int b = idx / NPTS, n = idx % NPTS;
  const float* xb = x + (size_t)b*3*NPTS + n;
  float x0 = xb[0], x1 = xb[NPTS], x2 = xb[2*NPTS];
  const float* T = trans + b*9;                    // T[c][k]
  float t0 = x0*T[0] + x1*T[3] + x2*T[6];
  float t1 = x0*T[1] + x1*T[4] + x2*T[7];
  float t2 = x0*T[2] + x1*T[5] + x2*T[8];
  _Float16* out = h1out + (size_t)idx*64;
#pragma unroll 1
  for (int c = 0; c < 64; ++c) {
    float acc = w[c*3+0]*t0 + w[c*3+1]*t1 + w[c*3+2]*t2;
    out[c] = (_Float16)fmaxf(fmaf(acc, sc[c], sh[c]), 0.f);
  }
}

// ---- tiny helpers ----------------------------------------------------------
__global__ void cvt_f32_f16_kernel(const float* s, _Float16* d, int n) {
  int i = blockIdx.x*256 + threadIdx.x;
  if (i < n) d[i] = (_Float16)s[i];
}
// fold conv bias + BN into per-channel scale/shift: y = sc*acc + sh
__global__ void fold_bn_kernel(const float* g, const float* b, const float* m,
                               const float* v, const float* bias,
                               float* sc, float* sh, int n) {
  int i = blockIdx.x*256 + threadIdx.x;
  if (i < n) {
    float s = g[i] * rsqrtf(v[i] + EPS_BN);
    sc[i] = s;
    sh[i] = s * (bias[i] - m[i]) + b[i];
  }
}
__global__ void finalize_z_kernel(const unsigned* m, float* out, int n) {
  int i = blockIdx.x*256 + threadIdx.x;
  if (i < n) out[i] = ord2f(m[i]);
}

// ---- host orchestration ----------------------------------------------------
extern "C" void kernel_launch(void* const* d_in, const int* in_sizes, int n_in,
                              void* d_out, int out_size, void* d_ws, size_t ws_size,
                              hipStream_t stream) {
  (void)in_sizes; (void)n_in; (void)out_size; (void)ws_size;
  // Recursive insertion-order flattening of setup_inputs():
  // 0:x, 1..32:stn{c1..f3 (w,b), bn1..bn5 (g,b,m,v)}, 33..64:fstn{...},
  // 65..70:c1/c2/c3 (w,b), 71..82:bn1/bn2/bn3, 83:valid_pts
  enum { S_C1W=0,S_C1B,S_C2W,S_C2B,S_C3W,S_C3B,S_F1W,S_F1B,S_F2W,S_F2B,S_F3W,S_F3B,
         S_BN1=12,S_BN2=16,S_BN3=20,S_BN4=24,S_BN5=28 };
  const int I_X=0, I_STN=1, I_FSTN=33, I_C1W=65, I_C1B=66, I_C2W=67, I_C2B=68,
            I_C3W=69, I_C3B=70, I_BN1=71, I_BN2=75, I_BN3=79, I_VP=83;
  auto Fp = [&](int i){ return (const float*)d_in[i]; };

  float* out       = (float*)d_out;
  float* out_z     = out;               // 32*1024
  float* out_trans = out + 32768;       // 32*9
  float* out_tfeat = out + 33056;       // 32*4096

  char* ws = (char*)d_ws;
  _Float16* h1       = (_Float16*)ws;                              // 16 MB
  unsigned* max_stn  = (unsigned*)(ws + (size_t)BATCH*NPTS*64*2);
  unsigned* max_fstn = max_stn  + BATCH*1024;
  unsigned* max_main = max_fstn + BATCH*1024;
  _Float16* w16        = (_Float16*)(max_main + BATCH*1024);
  _Float16* stn_c2w16  = w16;
  _Float16* stn_c3w16  = stn_c2w16  + 8192;
  _Float16* fstn_c1w16 = stn_c3w16  + 131072;
  _Float16* fstn_c2w16 = fstn_c1w16 + 4096;
  _Float16* fstn_c3w16 = fstn_c2w16 + 8192;
  _Float16* m_c2w16    = fstn_c3w16 + 131072;
  _Float16* m_c3w16    = m_c2w16    + 8192;
  float* fold = (float*)(m_c3w16 + 131072);   // 9 layers x (sc,sh)
  float* sSC[9]; float* sSH[9];
  { int sz[9] = {64,128,1024, 64,128,1024, 64,128,1024};
    float* p = fold;
    for (int i = 0; i < 9; ++i) { sSC[i] = p; p += sz[i]; sSH[i] = p; p += sz[i]; } }

  (void)hipMemsetAsync(max_stn, 0, (size_t)3*BATCH*1024*4, stream); // ordered-uint max identity
  auto cvt = [&](const float* s, _Float16* d, int n){
    cvt_f32_f16_kernel<<<(n+255)/256, 256, 0, stream>>>(s, d, n);
  };
  cvt(Fp(I_STN +S_C2W), stn_c2w16,  8192);
  cvt(Fp(I_STN +S_C3W), stn_c3w16,  131072);
  cvt(Fp(I_FSTN+S_C1W), fstn_c1w16, 4096);
  cvt(Fp(I_FSTN+S_C2W), fstn_c2w16, 8192);
  cvt(Fp(I_FSTN+S_C3W), fstn_c3w16, 131072);
  cvt(Fp(I_C2W),        m_c2w16,    8192);
  cvt(Fp(I_C3W),        m_c3w16,    131072);

  auto fold_bn = [&](int bnBase, int biasIdx, int slot, int n){
    fold_bn_kernel<<<(n+255)/256, 256, 0, stream>>>(
        Fp(bnBase), Fp(bnBase+1), Fp(bnBase+2), Fp(bnBase+3), Fp(biasIdx),
        sSC[slot], sSH[slot], n);
  };
  fold_bn(I_STN +S_BN1, I_STN +S_C1B, 0, 64);
  fold_bn(I_STN +S_BN2, I_STN +S_C2B, 1, 128);
  fold_bn(I_STN +S_BN3, I_STN +S_C3B, 2, 1024);
  fold_bn(I_FSTN+S_BN1, I_FSTN+S_C1B, 3, 64);
  fold_bn(I_FSTN+S_BN2, I_FSTN+S_C2B, 4, 128);
  fold_bn(I_FSTN+S_BN3, I_FSTN+S_C3B, 5, 1024);
  fold_bn(I_BN1, I_C1B, 6, 64);
  fold_bn(I_BN2, I_C2B, 7, 128);
  fold_bn(I_BN3, I_C3B, 8, 1024);

  dim3 gconv(BATCH*(NPTS/NT));   // 1024 workgroups

  Stn3Args sa{ Fp(I_X), Fp(I_STN+S_C1W), sSC[0], sSH[0],
               stn_c2w16, sSC[1], sSH[1], stn_c3w16, sSC[2], sSH[2], max_stn };
  stn3_conv_kernel<<<gconv, TPB, 0, stream>>>(sa);

  FcArgs fa1{ max_stn, Fp(I_STN+S_F1W), Fp(I_STN+S_F1B), Fp(I_STN+S_F2W), Fp(I_STN+S_F2B),
              Fp(I_STN+S_F3W), Fp(I_STN+S_F3B),
              Fp(I_STN+S_BN4), Fp(I_STN+S_BN4+1), Fp(I_STN+S_BN4+2), Fp(I_STN+S_BN4+3),
              Fp(I_STN+S_BN5), Fp(I_STN+S_BN5+1), Fp(I_STN+S_BN5+2), Fp(I_STN+S_BN5+3),
              out_trans };
  stn_fc_kernel<3><<<BATCH, 256, 0, stream>>>(fa1);

  apply3_c1_kernel<<<(BATCH*NPTS)/256, 256, 0, stream>>>(
      Fp(I_X), out_trans, Fp(I_C1W), sSC[6], sSH[6], h1);

  FstnArgs ff{ h1, fstn_c1w16, sSC[3], sSH[3], fstn_c2w16, sSC[4], sSH[4],
               fstn_c3w16, sSC[5], sSH[5], max_fstn };
  fstn_conv_kernel<<<gconv, TPB, 0, stream>>>(ff);

  FcArgs fa2{ max_fstn, Fp(I_FSTN+S_F1W), Fp(I_FSTN+S_F1B), Fp(I_FSTN+S_F2W), Fp(I_FSTN+S_F2B),
              Fp(I_FSTN+S_F3W), Fp(I_FSTN+S_F3B),
              Fp(I_FSTN+S_BN4), Fp(I_FSTN+S_BN4+1), Fp(I_FSTN+S_BN4+2), Fp(I_FSTN+S_BN4+3),
              Fp(I_FSTN+S_BN5), Fp(I_FSTN+S_BN5+1), Fp(I_FSTN+S_BN5+2), Fp(I_FSTN+S_BN5+3),
              out_tfeat };
  stn_fc_kernel<64><<<BATCH, 256, 0, stream>>>(fa2);

  MainArgs ma{ h1, out_tfeat, m_c2w16, sSC[7], sSH[7], m_c3w16, sSC[8], sSH[8],
               (const int*)d_in[I_VP], max_main };
  main_conv_kernel<<<gconv, TPB, 0, stream>>>(ma);

  finalize_z_kernel<<<(BATCH*1024)/256, 256, 0, stream>>>(max_main, out_z, BATCH*1024);
}
